// CRF_BiLSTM_7103875907720
// MI455X (gfx1250) — matrix operand order repaired and checked by reading, not compiled, not run on hardware
//
#include <hip/hip_runtime.h>
#include <hip/hip_bf16.h>
#include <math.h>

typedef __attribute__((ext_vector_type(16))) __bf16 v16bf;
typedef __attribute__((ext_vector_type(8)))  __bf16 v8bf;
typedef __attribute__((ext_vector_type(8)))  float  v8f;
typedef unsigned int u32x4 __attribute__((ext_vector_type(4)));
typedef int          i32x4v __attribute__((ext_vector_type(4)));
typedef int          i32x8v __attribute__((ext_vector_type(8)));

#define T_LEN 4096
#define HID   1024
#define G4    4096        // 4*HID
#define KTAG  50
#define START_TAG 48
#define END_TAG   49
#define NEGINF (-10000.0f)
#define NWG   16          // workgroups per direction in recurrence
#define NTILES 256        // G4/16 output tiles per direction
#define NCHUNK 32         // HID/32 K-chunks

// ---------------------------------------------------------------------------
// fp32 -> bf16 conversion
// ---------------------------------------------------------------------------
__global__ void cvt_bf16_kernel(const float* __restrict__ src,
                                __bf16* __restrict__ dst, int n) {
  int i = blockIdx.x * blockDim.x + threadIdx.x;
  if (i < n) dst[i] = (__bf16)src[i];
}

// ---------------------------------------------------------------------------
// WMMA bf16 GEMM:  XG[T,4H] = X[T,H] * W[4H,H]^T + (b_ih + b_hh)
// ---------------------------------------------------------------------------
__device__ inline v16bf load_a_frag(const __bf16* __restrict__ A, int lda,
                                    int m0, int kk, int lane) {
  // 16-bit A 16x32: lane&15 = M; lanes<16 hold K 0-15; lanes>=16 K 16-31
  int m    = m0 + (lane & 15);
  int koff = (lane >> 4) * 16;
  const __bf16* p = A + (size_t)m * lda + kk + koff;
  v8bf lo = *(const v8bf*)p;
  v8bf hi = *(const v8bf*)(p + 8);
  v16bf a;
#pragma unroll
  for (int e = 0; e < 8; ++e) { a[e] = lo[e]; a[8 + e] = hi[e]; }
  return a;
}

__device__ inline v16bf load_b_frag(const __bf16* __restrict__ W, int ldw,
                                    int n0, int kk, int lane) {
  // 16-bit B 32x16: lane&15 = N; lanes<16: K 0-15, lanes>=16: K 16-31
  int n  = n0 + (lane & 15);
  int kh = (lane >> 4) * 16;
  const __bf16* p = W + (size_t)n * ldw + kk + kh;
  return *(const v16bf*)p;
}

__global__ void gemm_xg_kernel(const __bf16* __restrict__ X,    // [T,H]
                               const __bf16* __restrict__ W,    // [4H,H]
                               const float*  __restrict__ b_ih,
                               const float*  __restrict__ b_hh,
                               float* __restrict__ XG) {        // [T,4H]
  int lane = threadIdx.x & 31;
  int wv   = threadIdx.x >> 5;
  int m0   = blockIdx.x * 16;
  int n0   = (blockIdx.y * 8 + wv) * 64;

  v8f acc[4];
#pragma unroll
  for (int i = 0; i < 4; ++i) acc[i] = (v8f)(0.0f);

  for (int kk = 0; kk < HID; kk += 32) {
    v16bf a = load_a_frag(X, HID, m0, kk, lane);
#pragma unroll
    for (int i = 0; i < 4; ++i) {
      v16bf b = load_b_frag(W, HID, n0 + 16 * i, kk, lane);
      acc[i] = __builtin_amdgcn_wmma_f32_16x16x32_bf16(
          false, a, false, b, (short)0, acc[i], false, false);
    }
  }

  int mbase = m0 + ((lane >> 4) * 8);
#pragma unroll
  for (int i = 0; i < 4; ++i) {
    int n = n0 + 16 * i + (lane & 15);
    float bias = b_ih[n] + b_hh[n];
#pragma unroll
    for (int r = 0; r < 8; ++r)
      XG[(size_t)(mbase + r) * G4 + n] = acc[i][r] + bias;
  }
}

// ---------------------------------------------------------------------------
// Pre-pack W_hh (fp32 [4H,H]) into WMMA B-fragment layout, gate-permuted so
// workgroup `wg` owns tiles [wg*16, wg*16+16) covering row_local = gate*64+jl
// with global row = gate*HID + wg*64 + jl.
// ---------------------------------------------------------------------------
__global__ void prepack_whh_kernel(const float* __restrict__ whh,
                                   __bf16* __restrict__ packed) {
  int gidx = blockIdx.x * blockDim.x + threadIdx.x;   // tile*1024 + chunk*32 + lane
  if (gidx >= NTILES * NCHUNK * 32) return;
  int lane  = gidx & 31;
  int chunk = (gidx >> 5) & 31;
  int tile  = gidx >> 10;
  int wg = tile >> 4;
  int tt = tile & 15;
  int row_local = tt * 16 + (lane & 15);
  int gate = row_local >> 6;
  int jl   = row_local & 63;
  int grow = gate * HID + wg * 64 + jl;
  int kbase = chunk * 32 + (lane >> 4) * 16;
  __bf16* dst = packed + (size_t)gidx * 16;
  const float* src = whh + (size_t)grow * HID + kbase;
#pragma unroll
  for (int e = 0; e < 16; ++e) dst[e] = (__bf16)src[e];
}

// ---------------------------------------------------------------------------
// TDM: DMA one row of `nelem` 2-byte elements from global to LDS.
// D# per CDNA5 ISA (groups 2/3 zero => <=2D tensor).
// 6-arg builtin form (clang-23 / therock-10.0 headers).
// ---------------------------------------------------------------------------
__device__ inline void tdm_load_row_to_lds(unsigned int lds_addr,
                                           const void* gptr,
                                           unsigned int nelem) {
  unsigned long long ga = (unsigned long long)(uintptr_t)gptr;
  u32x4 g0;
  g0[0] = 1u;                                            // count=1 (valid user D#)
  g0[1] = lds_addr;                                      // LDS byte address
  g0[2] = (unsigned int)(ga & 0xFFFFFFFFu);              // global_addr[31:0]
  g0[3] = (unsigned int)((ga >> 32) & 0x1FFFFFFu) | (2u << 30);  // [56:32] | type=2
  i32x8v g1;
  g1[0] = (int)(1u << 16);                               // data_size=1 (2B), mask=0
  g1[1] = (int)((nelem & 0xFFFFu) << 16);                // tensor_dim0[15:0] @48
  g1[2] = (int)((nelem >> 16) | (1u << 16));             // dim0[31:16], tensor_dim1=1
  g1[3] = (int)((nelem & 0xFFFFu) << 16);                // tile_dim0 @112
  g1[4] = 1;                                             // tile_dim1=1, tile_dim2=0
  g1[5] = (int)nelem;                                    // tensor_dim0_stride[31:0]
  g1[6] = 0;
  g1[7] = 0;
  i32x4v z4 = {0, 0, 0, 0};
  i32x8v z8 = {0, 0, 0, 0, 0, 0, 0, 0};
  __builtin_amdgcn_tensor_load_to_lds(g0, g1, z4, z4, z8, 0);
}

// ---------------------------------------------------------------------------
// Persistent bidirectional LSTM recurrence, WMMA GEMV (M padded to 16).
// h_prev staged into LDS via the Tensor Data Mover each timestep.
// ---------------------------------------------------------------------------
__global__ void lstm_recurrence_kernel(const __bf16* __restrict__ packed_f,
                                       const __bf16* __restrict__ packed_b,
                                       const float*  __restrict__ xg_f,
                                       const float*  __restrict__ xg_b,
                                       float* __restrict__ h_f,
                                       float* __restrict__ h_b,
                                       __bf16* __restrict__ hbf_f,
                                       __bf16* __restrict__ hbf_b,
                                       unsigned* __restrict__ sync) {
  int dir = blockIdx.x / NWG;            // 0 = forward, 1 = backward
  int wg  = blockIdx.x % NWG;
  const __bf16* P   = dir ? packed_b : packed_f;
  const float*  XG  = dir ? xg_b  : xg_f;
  float*       Hout = dir ? h_b   : h_f;
  __bf16*      HbfG = dir ? hbf_b : hbf_f;
  volatile unsigned* ctr = (volatile unsigned*)(sync + dir);

  __shared__ __bf16 hbf[HID];            // h_prev as bf16 (TDM destination)
  __shared__ float gbuf[256];            // gate pre-activations (row_local)
  __shared__ float c_lds[64];

  int tid  = threadIdx.x;
  int lane = tid & 31;
  int wv   = tid >> 5;                   // wave 0..7, owns tiles 2wv, 2wv+1
  const __bf16* ptile0 = P + (size_t)(wg * 16 + 2 * wv) * NCHUNK * 512;
  const __bf16* ptile1 = ptile0 + (size_t)NCHUNK * 512;
  int  laneoff = lane * 16;
  bool arow    = (lane & 15) == 0;       // lanes carrying A row 0
  int  aoff    = (lane >> 4) * 16;
  int my_gate = tid >> 6, my_jl = tid & 63;
  int my_row  = my_gate * HID + wg * 64 + my_jl;
  unsigned int hbf_lds_addr = (unsigned int)(uintptr_t)&hbf[0];

  if (tid < 64) c_lds[tid] = 0.0f;

  for (int it = 0; it < T_LEN; ++it) {
    int t = dir ? (T_LEN - 1 - it) : it;

    // prefetch next step's xg row into cache while we wait
    if (it + 1 < T_LEN) {
      int tn = dir ? (t - 1) : (t + 1);
      __builtin_prefetch(&XG[(size_t)tn * G4 + my_row], 0, 0);
    }

    if (it > 0) {
      if (tid == 0) {
        unsigned target = (unsigned)(NWG * it);
        while (*ctr < target) __builtin_amdgcn_s_sleep(2);
      }
      __syncthreads();
      __threadfence();
      int tp = dir ? (t + 1) : (t - 1);
      if (wv == 0) {   // one wave DMAs h_prev (bf16) into LDS via the TDM
        tdm_load_row_to_lds(hbf_lds_addr, HbfG + (size_t)tp * HID, HID);
        __builtin_amdgcn_s_wait_tensorcnt(0);
      }
    } else {
      for (int k = tid; k < HID; k += 256) hbf[k] = (__bf16)0.0f;
    }
    __syncthreads();

    v8f acc0 = (v8f)(0.0f), acc1 = (v8f)(0.0f);
#pragma unroll 4
    for (int c = 0; c < NCHUNK; ++c) {
      int kk = c * 32;
      v8bf lo = *(const v8bf*)&hbf[kk + aoff];        // K kk+aoff..+7
      v8bf hi = *(const v8bf*)&hbf[kk + 8 + aoff];    // K kk+aoff+8..+15
      v16bf a;
#pragma unroll
      for (int e = 0; e < 8; ++e) {
        a[e]     = arow ? lo[e] : (__bf16)0.0f;
        a[8 + e] = arow ? hi[e] : (__bf16)0.0f;
      }
      v16bf b0 = *(const v16bf*)(ptile0 + (size_t)c * 512 + laneoff);
      v16bf b1 = *(const v16bf*)(ptile1 + (size_t)c * 512 + laneoff);
      acc0 = __builtin_amdgcn_wmma_f32_16x16x32_bf16(
          false, a, false, b0, (short)0, acc0, false, false);
      acc1 = __builtin_amdgcn_wmma_f32_16x16x32_bf16(
          false, a, false, b1, (short)0, acc1, false, false);
    }
    // D row 0 (M=0) lives in lanes 0-15 of VGPR0: gate value for N = lane
    if (lane < 16) {
      gbuf[(2 * wv) * 16 + lane]     = acc0[0];
      gbuf[(2 * wv + 1) * 16 + lane] = acc1[0];
    }
    __syncthreads();

    if (tid < 64) {
      const float* xr = &XG[(size_t)t * G4 + wg * 64 + tid];
      float gi = gbuf[tid]       + xr[0 * HID];
      float gf = gbuf[64 + tid]  + xr[1 * HID];
      float gg = gbuf[128 + tid] + xr[2 * HID];
      float go = gbuf[192 + tid] + xr[3 * HID];
      float i_ = 1.0f / (1.0f + __expf(-gi));
      float f_ = 1.0f / (1.0f + __expf(-gf));
      float g_ = tanhf(gg);
      float o_ = 1.0f / (1.0f + __expf(-go));
      float c = f_ * c_lds[tid] + i_ * g_;
      c_lds[tid] = c;
      float h = o_ * tanhf(c);
      Hout[(size_t)t * HID + wg * 64 + tid] = h;         // fp32 for dense layer
      HbfG[(size_t)t * HID + wg * 64 + tid] = (__bf16)h; // bf16 for next TDM stage
    }
    __syncthreads();
    if (tid == 0) {
      __threadfence();
      atomicAdd(sync + dir, 1u);
    }
  }
}

// ---------------------------------------------------------------------------
// Dense projection: feats[t,k] = [h_f(t) ; h_b(t)] . dense_w[k,:] + b[k]
// ---------------------------------------------------------------------------
__global__ void dense_feats_kernel(const float* __restrict__ h_f,
                                   const float* __restrict__ h_b,
                                   const float* __restrict__ Wd,
                                   const float* __restrict__ bd,
                                   float* __restrict__ feats) {
  int t = blockIdx.x;
  int k = threadIdx.x;
  if (k >= KTAG) return;
  const float* wr = Wd + (size_t)k * (2 * HID);
  const float* hf = h_f + (size_t)t * HID;
  const float* hb = h_b + (size_t)t * HID;
  float acc = bd[k];
#pragma unroll 4
  for (int jj = 0; jj < HID; ++jj) acc += hf[jj] * wr[jj];
#pragma unroll 4
  for (int jj = 0; jj < HID; ++jj) acc += hb[jj] * wr[HID + jj];
  feats[(size_t)t * KTAG + k] = acc;
}

// ---------------------------------------------------------------------------
// Viterbi forward + backtrack (single workgroup, transitions in LDS).
// out[0] = path score; out[1..T] = best path tags (as float).
// ---------------------------------------------------------------------------
__global__ void viterbi_kernel(const float* __restrict__ feats,
                               const float* __restrict__ trans,
                               int* __restrict__ bptrs,
                               float* __restrict__ out) {
  __shared__ float trs[KTAG * KTAG];
  __shared__ float fv[64];
  __shared__ float fv2[64];
  int tid = threadIdx.x;

  for (int i = tid; i < KTAG * KTAG; i += 64) trs[i] = trans[i];
  if (tid < KTAG) fv[tid] = (tid == START_TAG) ? 0.0f : NEGINF;
  __syncthreads();

  for (int t = 0; t < T_LEN; ++t) {
    if (tid < KTAG) {
      const float* tr = trs + tid * KTAG;
      float best = -3.4e38f; int bi = 0;
      for (int p = 0; p < KTAG; ++p) {
        float s = tr[p] + fv[p];
        if (s > best) { best = s; bi = p; }
      }
      fv2[tid] = best + feats[(size_t)t * KTAG + tid];
      bptrs[(size_t)t * KTAG + tid] = bi;
    }
    __syncthreads();
    if (tid < KTAG) fv[tid] = fv2[tid];
    __syncthreads();
  }

  if (tid == 0) {
    float best = -3.4e38f; int bt = 0;
    for (int k = 0; k < KTAG; ++k) {
      float s = fv[k] + trs[END_TAG * KTAG + k];
      if (s > best) { best = s; bt = k; }
    }
    out[0] = best;
    int tag = bt;
    out[1 + (T_LEN - 1)] = (float)tag;
    for (int t = T_LEN - 1; t >= 1; --t) {
      tag = bptrs[(size_t)t * KTAG + tag];
      out[t] = (float)tag;
    }
  }
}

// ---------------------------------------------------------------------------
// Host orchestration
// ---------------------------------------------------------------------------
extern "C" void kernel_launch(void* const* d_in, const int* in_sizes, int n_in,
                              void* d_out, int out_size, void* d_ws, size_t ws_size,
                              hipStream_t stream) {
  const float* sentence = (const float*)d_in[0];
  const float* w_ih_f   = (const float*)d_in[1];
  const float* w_hh_f   = (const float*)d_in[2];
  const float* b_ih_f   = (const float*)d_in[3];
  const float* b_hh_f   = (const float*)d_in[4];
  const float* w_ih_b   = (const float*)d_in[5];
  const float* w_hh_b   = (const float*)d_in[6];
  const float* b_ih_b   = (const float*)d_in[7];
  const float* b_hh_b   = (const float*)d_in[8];
  const float* dense_w  = (const float*)d_in[9];
  const float* dense_b  = (const float*)d_in[10];
  const float* trans    = (const float*)d_in[11];
  float* out = (float*)d_out;

  char* base = (char*)d_ws;
  size_t off = 0;
  auto carve = [&](size_t bytes) -> char* {
    char* p = base + off;
    off += (bytes + 255) & ~(size_t)255;
    return p;
  };
  __bf16* x_bf     = (__bf16*)carve((size_t)T_LEN * HID * 2);
  __bf16* wih_f_bf = (__bf16*)carve((size_t)G4 * HID * 2);
  __bf16* wih_b_bf = (__bf16*)carve((size_t)G4 * HID * 2);
  __bf16* pack_f   = (__bf16*)carve((size_t)G4 * HID * 2);   // fragment-packed W_hh_f
  __bf16* pack_b   = (__bf16*)carve((size_t)G4 * HID * 2);   // fragment-packed W_hh_b
  float*  xg_f     = (float*)carve((size_t)T_LEN * G4 * 4);
  float*  xg_b     = (float*)carve((size_t)T_LEN * G4 * 4);
  float*  h_f      = (float*)carve((size_t)T_LEN * HID * 4);
  float*  h_b      = (float*)carve((size_t)T_LEN * HID * 4);
  __bf16* hbf_f    = (__bf16*)carve((size_t)T_LEN * HID * 2);
  __bf16* hbf_b    = (__bf16*)carve((size_t)T_LEN * HID * 2);
  float*  feats    = (float*)carve((size_t)T_LEN * KTAG * 4);
  int*    bptrs    = (int*)carve((size_t)T_LEN * KTAG * 4);
  unsigned* sync   = (unsigned*)carve(256);

  (void)hipMemsetAsync(sync, 0, 256, stream);

  // fp32 -> bf16 conversions (activations + input weights)
  {
    int n1 = T_LEN * HID;
    int n2 = G4 * HID;
    cvt_bf16_kernel<<<(n1 + 255) / 256, 256, 0, stream>>>(sentence, x_bf, n1);
    cvt_bf16_kernel<<<(n2 + 255) / 256, 256, 0, stream>>>(w_ih_f, wih_f_bf, n2);
    cvt_bf16_kernel<<<(n2 + 255) / 256, 256, 0, stream>>>(w_ih_b, wih_b_bf, n2);
  }

  // Fragment-pack recurrent weights
  {
    int nthr = NTILES * NCHUNK * 32;
    prepack_whh_kernel<<<(nthr + 255) / 256, 256, 0, stream>>>(w_hh_f, pack_f);
    prepack_whh_kernel<<<(nthr + 255) / 256, 256, 0, stream>>>(w_hh_b, pack_b);
  }

  // Input-gate GEMMs (WMMA bf16)
  {
    dim3 grid(T_LEN / 16, G4 / (8 * 64));
    gemm_xg_kernel<<<grid, 256, 0, stream>>>(x_bf, wih_f_bf, b_ih_f, b_hh_f, xg_f);
    gemm_xg_kernel<<<grid, 256, 0, stream>>>(x_bf, wih_b_bf, b_ih_b, b_hh_b, xg_b);
  }

  // Bidirectional recurrence: WMMA GEMV per step + TDM h-staging
  lstm_recurrence_kernel<<<2 * NWG, 256, 0, stream>>>(
      pack_f, pack_b, xg_f, xg_b, h_f, h_b, hbf_f, hbf_b, sync);

  // Dense projection
  dense_feats_kernel<<<T_LEN, 64, 0, stream>>>(h_f, h_b, dense_w, dense_b, feats);

  // Viterbi decode
  viterbi_kernel<<<1, 64, 0, stream>>>(feats, trans, bptrs, out);
}
